// BinConv2d_71674414236179
// MI455X (gfx1250) — matrix operand order repaired and checked by reading, not compiled, not run on hardware
//
#include <hip/hip_runtime.h>
#include <hip/hip_bf16.h>

typedef __attribute__((ext_vector_type(16))) __bf16       v16bf;
typedef __attribute__((ext_vector_type(8)))  float        v8f;
typedef __attribute__((ext_vector_type(4)))  unsigned int v4u;
typedef __attribute__((ext_vector_type(8)))  int          v8i;
typedef __attribute__((ext_vector_type(4)))  int          v4i;

#define CIN      256
#define COUT     256
#define HW       3136
#define IMGW     56
#define NTAPS    9

#define CO_TILE  128
#define PX_TILE  64
#define CI_CHUNK 32
#define AROW     40      // padded A row (ushorts): 32 ci + 8 pad -> 80B (64B data + 16B TDM pad)
#define XROW     40      // padded X row (ushorts)
#define XROWS    179     // 178 halo'd pixels + 1 zero row

#if __has_builtin(__builtin_amdgcn_tensor_load_to_lds) && \
    __has_builtin(__builtin_amdgcn_s_wait_tensorcnt)
#define USE_TDM 1
#else
#define USE_TDM 0
#endif

union FragU { v4u u4[2]; v16bf bf; };

__device__ __forceinline__ unsigned short f2bf_rn(float f) {
  unsigned int u = __builtin_bit_cast(unsigned int, f);
  unsigned int r = u + 0x7FFFu + ((u >> 16) & 1u);   // round-to-nearest-even
  return (unsigned short)(r >> 16);
}

// sign(W) -> bf16 bit patterns, layout [tap][co][ci]
__global__ void binconv_prep_sign_bf16(const float* __restrict__ W,
                                       unsigned short* __restrict__ Wb) {
  int idx = blockIdx.x * blockDim.x + threadIdx.x;
  if (idx >= NTAPS * COUT * CIN) return;
  int ci = idx & 255;
  int co = (idx >> 8) & 255;
  int t  = idx >> 16;
  float w = W[(co * CIN + ci) * NTAPS + t];
  Wb[idx] = (w > 0.f) ? 0x3F80u : ((w < 0.f) ? 0xBF80u : 0u);
}

__global__ void __launch_bounds__(256, 2)
binconv_wmma(const float* __restrict__ x,
             const unsigned short* __restrict__ Wb,
             float* __restrict__ out) {
  __shared__ __align__(16) unsigned short A_sh[NTAPS * CO_TILE * AROW]; // 92.2 KB
  __shared__ __align__(16) unsigned short Xhi[XROWS * XROW];            // 14.3 KB
  __shared__ __align__(16) unsigned short Xlo[XROWS * XROW];            // 14.3 KB

  const int tid   = threadIdx.x;
  const int lane  = tid & 31;
  const int lrow  = lane & 15;
  const int lhalf = lane >> 4;
  const int wv    = tid >> 5;
  const int wm    = wv & 3;    // co wave 0..3 (32 co each)
  const int wn    = wv >> 2;   // px wave 0..1 (32 px each)

  const int px0 = blockIdx.x * PX_TILE;
  const int co0 = blockIdx.y * CO_TILE;
  const int n   = blockIdx.z;

  // per-N-subtile pixel geometry (flattened h*56+w)
  int pl[2], ph[2], pw[2];
#pragma unroll
  for (int ni = 0; ni < 2; ++ni) {
    pl[ni] = wn * 32 + ni * 16 + lrow;      // 0..63 within tile
    int px = px0 + pl[ni];
    ph[ni] = px / IMGW;
    pw[ni] = px - ph[ni] * IMGW;
  }

  v8f acc[2][2];
#pragma unroll
  for (int mi = 0; mi < 2; ++mi)
#pragma unroll
    for (int ni = 0; ni < 2; ++ni)
#pragma unroll
      for (int k = 0; k < 8; ++k) acc[mi][ni][k] = 0.f;

  // zero row (index 178) used to realize zero-padding via pointer redirect
  if (tid < 32) { Xhi[178 * XROW + tid] = 0; Xlo[178 * XROW + tid] = 0; }

  const float* xbase = x + (size_t)n * CIN * HW;

  for (int ci0 = 0; ci0 < CIN; ci0 += CI_CHUNK) {
    __syncthreads();   // previous compute done reading A_sh / X planes

#if USE_TDM
    // ---- TDM: one 3D tensor_load_to_lds stages all 9 taps' weight tiles ----
    // Tile: x = 32 ci (64B row), y = 128 co, z = 9 taps.
    // LDS rows hardware-padded 64B -> 80B (pad_interval=16dw, pad_amount=4dw)
    // so the fill order matches A_sh[t][co][AROW] exactly.
    if (tid < 32) {    // wave 0 only (wave-uniform branch; TDM ignores EXEC)
      unsigned long long ga =
          (unsigned long long)(size_t)(Wb + ((size_t)co0 * CIN + ci0));
      unsigned lds_addr = (unsigned)(size_t)(&A_sh[0]);
      v4u g0 = { 1u,                               // count=1, is_restore=0
                 lds_addr,                         // lds_addr[31:0]
                 (unsigned)ga,                     // global_addr[31:0]
                 ((unsigned)(ga >> 32) & 0x01FFFFFFu) | 0x80000000u }; // type=2
      v8i g1;
      g1[0] = (int)((1u << 16)      // data_size = 2 bytes
                  | (1u << 20)      // pad_enable
                  | (3u << 22)      // pad_interval: 16 dwords (64B)
                  | (3u << 25));    // pad_amount:   4 dwords (16B)
      g1[1] = (int)(32u << 16);     // tensor_dim0[15:0]=32  (atomic_addr=0)
      g1[2] = (int)(128u << 16);    // tensor_dim0[31:16]=0, tensor_dim1[15:0]=128
      g1[3] = (int)(32u << 16);     // tensor_dim1[31:16]=0, tile_dim0=32
      g1[4] = (int)(128u | (9u << 16)); // tile_dim1=128, tile_dim2=9
      g1[5] = 256;                  // tensor_dim0_stride[31:0] = 256 elems
      g1[6] = 0;                    // dim0_stride[47:32]=0, dim1_stride[15:0]=0
      g1[7] = 1;                    // tensor_dim1_stride[47:16] = 65536>>16
      v4i g2 = { 9, 0, 0, 0 };      // tensor_dim2 = 9
      v4i g3 = { 0, 0, 0, 0 };
      v8i g4 = { 0, 0, 0, 0, 0, 0, 0, 0 };   // extra operand on 6-arg toolchain
      __builtin_amdgcn_tensor_load_to_lds(g0, g1, g2, g3, g4, 0);
    }
#endif

    // ---- stage activations: 32 ci x 178 px, f32 -> (hi,lo) bf16 planes ----
    for (int i = tid; i < CI_CHUNK * 178; i += 256) {
      int ci = i / 178;
      int p  = i - ci * 178;
      int px = px0 - 57 + p;                 // halo covers shift in [-57,57]
      float v = 0.f;
      if ((unsigned)px < (unsigned)HW) v = xbase[(size_t)(ci0 + ci) * HW + px];
      unsigned short hb = f2bf_rn(v);
      float hf = __builtin_bit_cast(float, (unsigned)hb << 16);
      unsigned short lb = f2bf_rn(v - hf);
      Xhi[p * XROW + ci] = hb;
      Xlo[p * XROW + ci] = lb;
    }

#if USE_TDM
    if (tid < 32) __builtin_amdgcn_s_wait_tensorcnt(0);
#else
    // ---- fallback: manual weight staging, 9 taps x 128 co x 32 ci ----
    {
      const v4u* __restrict__ src = (const v4u*)Wb;   // 8 ci per v4u
      for (int i = tid; i < NTAPS * CO_TILE * 4; i += 256) {
        int q   = i & 3;
        int row = i >> 2;                    // t*128 + co
        int t   = row >> 7;
        int co  = row & 127;
        v4u v = src[(((t * COUT + (co0 + co)) * CIN + ci0) >> 3) + q];
        *(v4u*)&A_sh[(t * CO_TILE + co) * AROW + q * 8] = v;
      }
    }
#endif
    __syncthreads();

#pragma unroll
    for (int t = 0; t < NTAPS; ++t) {
      const int dh = t / 3 - 1, dw = t % 3 - 1;
      const int shift = dh * IMGW + dw;

      // A fragments: 16x32 bf16 per ISA layout (dword K-pairs {8h..}, {16+8h..})
      FragU a[2];
#pragma unroll
      for (int mi = 0; mi < 2; ++mi) {
        const unsigned short* ap =
            &A_sh[(t * CO_TILE + wm * 32 + mi * 16 + lrow) * AROW];
        a[mi].u4[0] = *(const v4u*)(ap + 8 * lhalf);
        a[mi].u4[1] = *(const v4u*)(ap + 16 + 8 * lhalf);
      }

      // B fragments: 32x16 bf16 per ISA layout (K {8h..8h+7}, {16+8h..16+8h+7})
      FragU bh[2], bl[2];
#pragma unroll
      for (int ni = 0; ni < 2; ++ni) {
        bool valid = ((unsigned)(ph[ni] + dh) < 56u) &&
                     ((unsigned)(pw[ni] + dw) < 56u);
        int p = valid ? (pl[ni] + 57 + shift) : 178;   // redirect to zero row
        const unsigned short* bphi = &Xhi[p * XROW + 8 * lhalf];
        const unsigned short* bplo = &Xlo[p * XROW + 8 * lhalf];
        bh[ni].u4[0] = *(const v4u*)(bphi);
        bh[ni].u4[1] = *(const v4u*)(bphi + 16);
        bl[ni].u4[0] = *(const v4u*)(bplo);
        bl[ni].u4[1] = *(const v4u*)(bplo + 16);
      }

#pragma unroll
      for (int mi = 0; mi < 2; ++mi)
#pragma unroll
        for (int ni = 0; ni < 2; ++ni) {
          acc[mi][ni] = __builtin_amdgcn_wmma_f32_16x16x32_bf16(
              false, a[mi].bf, false, bh[ni].bf, (short)0, acc[mi][ni],
              false, false);
          acc[mi][ni] = __builtin_amdgcn_wmma_f32_16x16x32_bf16(
              false, a[mi].bf, false, bl[ni].bf, (short)0, acc[mi][ni],
              false, false);
        }
    }
  }

  // ---- epilogue: C/D layout — N = lane&15, M = r + 8*(lane>>4) ----
  float* obase = out + (size_t)n * COUT * HW;
#pragma unroll
  for (int mi = 0; mi < 2; ++mi)
#pragma unroll
    for (int ni = 0; ni < 2; ++ni) {
      int px = px0 + wn * 32 + ni * 16 + lrow;
#pragma unroll
      for (int r = 0; r < 8; ++r) {
        int co = co0 + wm * 32 + mi * 16 + r + 8 * lhalf;
        obase[(size_t)co * HW + px] = acc[mi][ni][r];
      }
    }
}

extern "C" void kernel_launch(void* const* d_in, const int* in_sizes, int n_in,
                              void* d_out, int out_size, void* d_ws, size_t ws_size,
                              hipStream_t stream) {
  const float* x = (const float*)d_in[0];   // (N,256,56,56) f32
  const float* W = (const float*)d_in[1];   // (256,256,3,3) f32 OIHW
  float* out = (float*)d_out;

  // workspace: sign(W) as bf16 bits, [9][256][256] = 1.18 MB
  unsigned short* Wb = (unsigned short*)d_ws;
  (void)ws_size; (void)n_in; (void)out_size;

  const int prep = NTAPS * COUT * CIN;
  binconv_prep_sign_bf16<<<(prep + 255) / 256, 256, 0, stream>>>(W, Wb);

  const int N = in_sizes[0] / (CIN * HW);   // 32
  dim3 grid(HW / PX_TILE, COUT / CO_TILE, N);   // (49, 2, N)
  binconv_wmma<<<grid, 256, 0, stream>>>(x, Wb, out);
}
